// LinearAttention_34789235098244
// MI455X (gfx1250) — matrix operand order repaired and checked by reading, not compiled, not run on hardware
//
#include <hip/hip_runtime.h>

// ---------------------------------------------------------------------------
// Linear attention (chunked) for MI455X / gfx1250 — bf16 WMMA pipeline.
//   K1: qkv = x @ wqkv  (bf16 WMMA, fp32 acc), fused ReLU(q,k), head-split
//       scatter to bf16 scratch (b,h,n,hd).
//   K2: per-(b,h) chunked causal linear attention; all GEMMs are
//       v_wmma_f32_16x16x32_bf16; KV state kept in WMMA accumulators.
//       Q-chunk staging uses global_load_async_to_lds_b128 (ASYNCcnt).
//   K3: out = y @ wo (bf16 WMMA), fp32 output; A staged via async-to-LDS.
// ---------------------------------------------------------------------------

typedef __attribute__((ext_vector_type(16))) __bf16 v16bf;
typedef __attribute__((ext_vector_type(8)))  __bf16 v8bf;
typedef __attribute__((ext_vector_type(8)))  float  v8f;
typedef __attribute__((ext_vector_type(4)))  float  v4f;

#define NHEADS 16
#define HD     64
#define SEQ    2048
#define DMODEL 1024
#define BATCH  2
#define CHUNK  64
#define NCHUNK (SEQ / CHUNK)
#define EPSV   1e-5f

// Native bf16 converts (v_cvt_bf16_f32 / v_cvt_pk_bf16_f32) instead of manual
// bit-twiddling: 1 VALU op per 1-2 elements vs 3 per element.
__device__ __forceinline__ unsigned short f2bf(float f) {
  union { __bf16 h; unsigned short u; } c;
  c.h = (__bf16)f;
  return c.u;
}
__device__ __forceinline__ unsigned pack2bf(float a, float b) {
  union { __bf16 h[2]; unsigned u; } c;
  c.h[0] = (__bf16)a;
  c.h[1] = (__bf16)b;
  return c.u;
}
__device__ __forceinline__ float bf2f(unsigned short h) {
  union { unsigned u; float f; } c;
  c.u = ((unsigned)h) << 16;
  return c.f;
}

// A-matrix fragment (16-bit 16x32): K contiguous, matching B: lanes 0-15 hold
// K = 0..15, lanes 16-31 hold K = 16..31.
// p points at row start + kbase + kh8 (kh8 = 8*(lane>>4)); load at p + kh8.
__device__ __forceinline__ v16bf make_frag_a(const unsigned short* p) {
  return *reinterpret_cast<const v16bf*>(p + ((threadIdx.x >> 4) & 1) * 8);
}
// B-matrix fragment (16-bit 32x16): lane holds one column, K contiguous:
// lanes 0-15 K=0..15, lanes 16-31 K=16..31 -> single 32B load.
__device__ __forceinline__ v16bf load_frag_b(const unsigned short* p) {
  return *reinterpret_cast<const v16bf*>(p);
}
__device__ __forceinline__ v8f wmma_bf16(v16bf a, v16bf b, v8f c) {
  return __builtin_amdgcn_wmma_f32_16x16x32_bf16(false, a, false, b, (short)0, c,
                                                 false, false);
}

// --- gfx1250 async global->LDS copy (tracked by ASYNCcnt) -------------------
typedef __attribute__((address_space(3))) unsigned short lds_u16_t;
__device__ __forceinline__ unsigned lds_off(const unsigned short* p) {
  // generic -> LDS addrspace, then ptrtoint: wave-relative LDS byte address
  return (unsigned)(unsigned long long)(lds_u16_t*)p;
}
__device__ __forceinline__ void async_copy_b128(unsigned lds_addr,
                                                const void* gaddr) {
  asm volatile("global_load_async_to_lds_b128 %0, %1, off"
               :
               : "v"(lds_addr), "v"((unsigned long long)gaddr)
               : "memory");
}
__device__ __forceinline__ void wait_async0() {
  asm volatile("s_wait_asynccnt 0" ::: "memory");
}

// ---------------------------------------------------------------------------
// Kernel 1: qkv GEMM (M=4096, K=1024, N=3072), 128x128 block tile, K-step 32.
// 256 threads = 8 waves; each wave computes 2x4 of the 8x8 16x16 tiles.
// Epilogue: ReLU on q,k regions; bf16 scatter to (b,h,n,hd) scratch.
// ---------------------------------------------------------------------------
__global__ __launch_bounds__(256) void qkv_gemm_relu_split(
    const float* __restrict__ x, const float* __restrict__ wqkv,
    unsigned short* __restrict__ qb, unsigned short* __restrict__ kb,
    unsigned short* __restrict__ vb) {
  __shared__ __align__(32) unsigned short As[128 * 32];  // [m][k]
  __shared__ __align__(32) unsigned short Bs[128 * 32];  // [n][k] (transposed)

  const int tid = threadIdx.x, lane = tid & 31, wave = tid >> 5;
  const int rowBase = blockIdx.y * 128, colBase = blockIdx.x * 128;
  const int wti = (wave >> 1) * 2, wtj = (wave & 1) * 4;
  const int rA = lane & 15;
  const int kh8 = (lane >> 4) * 8, kh16 = (lane >> 4) * 16;
  const int mrow = (lane >> 4) * 8;

  const v8f vzero = {0.f, 0.f, 0.f, 0.f, 0.f, 0.f, 0.f, 0.f};
  v8f acc[2][4];
#pragma unroll
  for (int i = 0; i < 2; ++i)
#pragma unroll
    for (int j = 0; j < 4; ++j) acc[i][j] = vzero;

  const int ra = tid >> 1, ca = (tid & 1) * 16;   // A stage: 16 fp32 each
  const int rb = tid >> 3, cb = (tid & 7) * 16;   // B stage: 16 fp32 each

  for (int k0 = 0; k0 < DMODEL; k0 += 32) {
    {  // stage A (fp32 -> bf16, packed u32 stores)
      const float* src = x + (size_t)(rowBase + ra) * DMODEL + k0 + ca;
      unsigned* dst = reinterpret_cast<unsigned*>(&As[ra * 32 + ca]);
#pragma unroll
      for (int i = 0; i < 16; i += 4) {
        v4f v = *reinterpret_cast<const v4f*>(src + i);
        dst[(i >> 1) + 0] = pack2bf(v[0], v[1]);
        dst[(i >> 1) + 1] = pack2bf(v[2], v[3]);
      }
    }
    {  // stage B transposed (fp32 -> bf16)
      const float* src = wqkv + (size_t)(k0 + rb) * (3 * DMODEL) + colBase + cb;
#pragma unroll
      for (int i = 0; i < 16; i += 4) {
        v4f v = *reinterpret_cast<const v4f*>(src + i);
        Bs[(cb + i + 0) * 32 + rb] = f2bf(v[0]);
        Bs[(cb + i + 1) * 32 + rb] = f2bf(v[1]);
        Bs[(cb + i + 2) * 32 + rb] = f2bf(v[2]);
        Bs[(cb + i + 3) * 32 + rb] = f2bf(v[3]);
      }
    }
    __syncthreads();
    if (k0 + 32 < DMODEL) {  // prefetch next K-slab (global_prefetch_b8)
      __builtin_prefetch(x + (size_t)(rowBase + ra) * DMODEL + k0 + 32 + ca, 0, 0);
      __builtin_prefetch(wqkv + (size_t)(k0 + 32 + rb) * (3 * DMODEL) + colBase + cb,
                         0, 0);
    }
    v16bf af[2];
#pragma unroll
    for (int i = 0; i < 2; ++i) {
      const int row = (wti + i) * 16 + rA;
      af[i] = make_frag_a(&As[row * 32 + kh8]);
    }
#pragma unroll
    for (int j = 0; j < 4; ++j) {
      const int col = (wtj + j) * 16 + rA;
      v16bf bf = load_frag_b(&Bs[col * 32 + kh16]);
#pragma unroll
      for (int i = 0; i < 2; ++i) acc[i][j] = wmma_bf16(af[i], bf, acc[i][j]);
    }
    __syncthreads();
  }

  // Epilogue: ReLU(q,k), bf16, scatter to (b*h, n, hd) layout.
  // `which`/head are uniform per column tile; per-e dst is base + e*HD.
#pragma unroll
  for (int j = 0; j < 4; ++j) {
    const int gcol = colBase + (wtj + j) * 16 + rA;  // 0..3071
    const int which = gcol >> 10, dcol = gcol & 1023;
    unsigned short* dstBuf = (which == 0) ? qb : (which == 1) ? kb : vb;
    const bool doRelu = (which < 2);
    const int hh = dcol >> 6, eh = dcol & 63;
#pragma unroll
    for (int i = 0; i < 2; ++i) {
      const int grow0 = rowBase + (wti + i) * 16 + mrow;  // multiple of 8
      const int bi = grow0 >> 11, ni0 = grow0 & 2047;     // e stays in-block
      const size_t base = ((size_t)(bi * NHEADS + hh) * SEQ + ni0) * HD + eh;
#pragma unroll
      for (int e = 0; e < 8; ++e) {
        float v = acc[i][j][e];
        if (doRelu) v = fmaxf(v, 0.0f);
        dstBuf[base + (size_t)e * HD] = f2bf(v);
      }
    }
  }
}

// ---------------------------------------------------------------------------
// Kernel 2: chunked causal linear attention, one block per (b,h).
// ---------------------------------------------------------------------------
__global__ __launch_bounds__(256) void linattn_chunked(
    const unsigned short* __restrict__ qb, const unsigned short* __restrict__ kb,
    const unsigned short* __restrict__ vb, unsigned short* __restrict__ yb) {
  __shared__ __align__(32) unsigned short Qs[CHUNK * HD];   // [t][d]
  __shared__ __align__(32) unsigned short Ks[CHUNK * HD];   // [s][d]
  __shared__ __align__(32) unsigned short Kts[HD * CHUNK];  // [d][s]
  __shared__ __align__(32) unsigned short Vts[HD * CHUNK];  // [e][s]
  __shared__ __align__(32) unsigned short Ss[CHUNK * CHUNK];// masked S [t][s]
  __shared__ __align__(32) unsigned short KVt[HD * HD];     // bf16 KV^T [e][d]
  __shared__ float Ksum[HD];
  __shared__ float den[CHUNK];

  const int bh = blockIdx.x;                 // 0..31
  const int bi = bh >> 4, h = bh & (NHEADS - 1);
  const unsigned short* Qg = qb + (size_t)bh * SEQ * HD;
  const unsigned short* Kg = kb + (size_t)bh * SEQ * HD;
  const unsigned short* Vg = vb + (size_t)bh * SEQ * HD;
  unsigned short* Yg = yb + (size_t)bi * SEQ * DMODEL + h * HD;

  const int tid = threadIdx.x, lane = tid & 31, wave = tid >> 5;
  const int ti = wave >> 1;                  // output tile row 0..3
  const int tj0 = (wave & 1) * 2;            // output tile cols {tj0, tj0+1}
  const int rA = lane & 15;
  const int kh8 = (lane >> 4) * 8, kh16 = (lane >> 4) * 16;
  const int mrow = (lane >> 4) * 8;

  const v8f vzero = {0.f, 0.f, 0.f, 0.f, 0.f, 0.f, 0.f, 0.f};
  v8f kvstate[2] = {vzero, vzero};           // fp32 KV state tiles in regs

  for (int i = tid; i < HD * HD; i += 256) KVt[i] = 0;
  if (tid < HD) Ksum[tid] = 0.0f;
  __syncthreads();

  const int rS = tid >> 2, cS = (tid & 3) * 16;  // staging: 16 elems/thread
  const unsigned qs_lds0 = lds_off(&Qs[rS * HD + cS]);
  const unsigned qs_lds1 = lds_off(&Qs[rS * HD + cS + 8]);

  for (int ch = 0; ch < NCHUNK; ++ch) {
    const int s0 = ch * CHUNK;
    const size_t off = (size_t)(s0 + rS) * HD + cS;
    // ---- stage Q via async global->LDS DMA (ASYNCcnt) ----
    async_copy_b128(qs_lds0, Qg + off);
    async_copy_b128(qs_lds1, Qg + off + 8);
    {  // ---- stage K (row-major + K^T) and V^T through registers ----
      union { uint4 v; unsigned short s[8]; } b0, b1, c0, c1;
      b0.v = reinterpret_cast<const uint4*>(Kg + off)[0];
      b1.v = reinterpret_cast<const uint4*>(Kg + off)[1];
      c0.v = reinterpret_cast<const uint4*>(Vg + off)[0];
      c1.v = reinterpret_cast<const uint4*>(Vg + off)[1];
      *reinterpret_cast<uint4*>(&Ks[rS * HD + cS]) = b0.v;
      *reinterpret_cast<uint4*>(&Ks[rS * HD + cS + 8]) = b1.v;
#pragma unroll
      for (int i = 0; i < 8; ++i) {
        Kts[(cS + i) * CHUNK + rS] = b0.s[i];
        Kts[(cS + 8 + i) * CHUNK + rS] = b1.s[i];
        Vts[(cS + i) * CHUNK + rS] = c0.s[i];
        Vts[(cS + 8 + i) * CHUNK + rS] = c1.s[i];
      }
    }
    wait_async0();
    __syncthreads();
    if (ch + 1 < NCHUNK) {  // prefetch next chunk
      const size_t noff = (size_t)(s0 + CHUNK + rS) * HD + cS;
      __builtin_prefetch(Qg + noff, 0, 0);
      __builtin_prefetch(Kg + noff, 0, 0);
      __builtin_prefetch(Vg + noff, 0, 0);
    }
    // ---- den_inter[t] = q_t . Ksum_prev ----
    if (tid < CHUNK) {
      float a = 0.0f;
#pragma unroll 8
      for (int d = 0; d < HD; ++d) a += bf2f(Qs[tid * HD + d]) * Ksum[d];
      den[tid] = a;
    }
    __syncthreads();
    // ---- S = Q K^T, causal mask, rowsum -> den (ds_add_f32 atomics) ----
#pragma unroll
    for (int jj = 0; jj < 2; ++jj) {
      const int tj = tj0 + jj;
      v8f s = vzero;
#pragma unroll
      for (int kk = 0; kk < HD; kk += 32) {
        v16bf a = make_frag_a(&Qs[(ti * 16 + rA) * HD + kk + kh8]);
        v16bf b = load_frag_b(&Ks[(tj * 16 + rA) * HD + kk + kh16]);
        s = wmma_bf16(a, b, s);
      }
#pragma unroll
      for (int e = 0; e < 8; ++e) {
        const int t = ti * 16 + mrow + e;
        const int sI = tj * 16 + rA;
        float v = (sI <= t) ? s[e] : 0.0f;
        float rs = v;  // reduce over the 16 lanes of this half (cols of row t)
        rs += __shfl_xor(rs, 1, 32);
        rs += __shfl_xor(rs, 2, 32);
        rs += __shfl_xor(rs, 4, 32);
        rs += __shfl_xor(rs, 8, 32);
        if (rA == 0) atomicAdd(&den[t], rs);
        Ss[t * CHUNK + sI] = f2bf(v);
      }
    }
    __syncthreads();
    // ---- num = S_mask @ V + Q @ KV_prev ; y = num/(den+eps) ----
#pragma unroll
    for (int jj = 0; jj < 2; ++jj) {
      const int te = tj0 + jj;
      v8f acc = vzero;
#pragma unroll
      for (int kk = 0; kk < CHUNK; kk += 32) {
        v16bf a = make_frag_a(&Ss[(ti * 16 + rA) * CHUNK + kk + kh8]);
        v16bf b = load_frag_b(&Vts[(te * 16 + rA) * CHUNK + kk + kh16]);
        acc = wmma_bf16(a, b, acc);
      }
#pragma unroll
      for (int kk = 0; kk < HD; kk += 32) {
        v16bf a = make_frag_a(&Qs[(ti * 16 + rA) * HD + kk + kh8]);
        v16bf b = load_frag_b(&KVt[(te * 16 + rA) * HD + kk + kh16]);
        acc = wmma_bf16(a, b, acc);
      }
#pragma unroll
      for (int e = 0; e < 8; ++e) {
        const int t = ti * 16 + mrow + e;
        const int ee = te * 16 + rA;
        const float y = acc[e] / (den[t] + EPSV);
        Yg[(size_t)(s0 + t) * DMODEL + ee] = f2bf(y);
      }
    }
    // ---- state: KV += K^T V, accumulated straight into WMMA registers ----
#pragma unroll
    for (int jj = 0; jj < 2; ++jj) {
      const int te = tj0 + jj;
#pragma unroll
      for (int kk = 0; kk < CHUNK; kk += 32) {
        v16bf a = make_frag_a(&Kts[(ti * 16 + rA) * CHUNK + kk + kh8]);
        v16bf b = load_frag_b(&Vts[(te * 16 + rA) * CHUNK + kk + kh16]);
        kvstate[jj] = wmma_bf16(a, b, kvstate[jj]);
      }
    }
    if (tid < HD) {  // Ksum += sum_s k_s
      float a = 0.0f;
#pragma unroll 8
      for (int s2 = 0; s2 < CHUNK; ++s2) a += bf2f(Kts[tid * CHUNK + s2]);
      Ksum[tid] += a;
    }
    __syncthreads();
    // ---- refresh bf16 transposed KV copy (packed u32 LDS stores) ----
#pragma unroll
    for (int jj = 0; jj < 2; ++jj) {
      const int te = tj0 + jj;
      const int d0 = ti * 16 + mrow;
      const int ee = te * 16 + rA;
      unsigned* dst = reinterpret_cast<unsigned*>(&KVt[ee * HD + d0]);
#pragma unroll
      for (int e = 0; e < 8; e += 2)
        dst[e >> 1] = pack2bf(kvstate[jj][e], kvstate[jj][e + 1]);
    }
    __syncthreads();
  }
}

// ---------------------------------------------------------------------------
// Kernel 3: out = y @ wo (M=4096, K=1024, N=1024), fp32 output.
// A tile staged via async global->LDS DMA.
// ---------------------------------------------------------------------------
__global__ __launch_bounds__(256) void out_gemm(
    const unsigned short* __restrict__ yb, const float* __restrict__ wo,
    float* __restrict__ out) {
  __shared__ __align__(32) unsigned short As[128 * 32];
  __shared__ __align__(32) unsigned short Bs[128 * 32];
  const int tid = threadIdx.x, lane = tid & 31, wave = tid >> 5;
  const int rowBase = blockIdx.y * 128, colBase = blockIdx.x * 128;
  const int wti = (wave >> 1) * 2, wtj = (wave & 1) * 4;
  const int rA = lane & 15;
  const int kh8 = (lane >> 4) * 8, kh16 = (lane >> 4) * 16;
  const int mrow = (lane >> 4) * 8;

  const v8f vzero = {0.f, 0.f, 0.f, 0.f, 0.f, 0.f, 0.f, 0.f};
  v8f acc[2][4];
#pragma unroll
  for (int i = 0; i < 2; ++i)
#pragma unroll
    for (int j = 0; j < 4; ++j) acc[i][j] = vzero;

  const int ra = tid >> 1, ca = (tid & 1) * 16;
  const int rb = tid >> 3, cb = (tid & 7) * 16;
  const unsigned as_lds0 = lds_off(&As[ra * 32 + ca]);
  const unsigned as_lds1 = lds_off(&As[ra * 32 + ca + 8]);

  for (int k0 = 0; k0 < DMODEL; k0 += 32) {
    // stage A (already bf16) via async global->LDS DMA
    const unsigned short* srcA = yb + (size_t)(rowBase + ra) * DMODEL + k0 + ca;
    async_copy_b128(as_lds0, srcA);
    async_copy_b128(as_lds1, srcA + 8);
    {  // stage B transposed (fp32 -> bf16)
      const float* src = wo + (size_t)(k0 + rb) * DMODEL + colBase + cb;
#pragma unroll
      for (int i = 0; i < 16; i += 4) {
        v4f v = *reinterpret_cast<const v4f*>(src + i);
        Bs[(cb + i + 0) * 32 + rb] = f2bf(v[0]);
        Bs[(cb + i + 1) * 32 + rb] = f2bf(v[1]);
        Bs[(cb + i + 2) * 32 + rb] = f2bf(v[2]);
        Bs[(cb + i + 3) * 32 + rb] = f2bf(v[3]);
      }
    }
    wait_async0();
    __syncthreads();
    if (k0 + 32 < DMODEL) {
      __builtin_prefetch(yb + (size_t)(rowBase + ra) * DMODEL + k0 + 32 + ca, 0, 0);
      __builtin_prefetch(wo + (size_t)(k0 + 32 + rb) * DMODEL + colBase + cb, 0, 0);
    }
    v16bf af[2];
#pragma unroll
    for (int i = 0; i < 2; ++i)
      af[i] = make_frag_a(&As[((wti + i) * 16 + rA) * 32 + kh8]);
#pragma unroll
    for (int j = 0; j < 4; ++j) {
      v16bf bf = load_frag_b(&Bs[((wtj + j) * 16 + rA) * 32 + kh16]);
#pragma unroll
      for (int i = 0; i < 2; ++i) acc[i][j] = wmma_bf16(af[i], bf, acc[i][j]);
    }
    __syncthreads();
  }
#pragma unroll
  for (int i = 0; i < 2; ++i)
#pragma unroll
    for (int j = 0; j < 4; ++j)
#pragma unroll
      for (int e = 0; e < 8; ++e) {
        const int m = (wti + i) * 16 + mrow + e;
        const int nn = (wtj + j) * 16 + rA;
        out[(size_t)(rowBase + m) * DMODEL + colBase + nn] = acc[i][j][e];
      }
}

// ---------------------------------------------------------------------------
extern "C" void kernel_launch(void* const* d_in, const int* in_sizes, int n_in,
                              void* d_out, int out_size, void* d_ws, size_t ws_size,
                              hipStream_t stream) {
  (void)in_sizes; (void)n_in; (void)out_size; (void)ws_size;
  const float* x    = (const float*)d_in[0];   // (2, 2048, 1024)
  const float* wqkv = (const float*)d_in[1];   // (1024, 3072)
  const float* wo   = (const float*)d_in[2];   // (1024, 1024)
  float* out = (float*)d_out;                  // (2, 2048, 1024)

  const size_t PH = (size_t)BATCH * NHEADS * SEQ * HD;  // 4,194,304 elems
  unsigned short* qb = (unsigned short*)d_ws;  // bf16 scratch, 4 x 8MB = 32MB
  unsigned short* kb = qb + PH;
  unsigned short* vb = kb + PH;
  unsigned short* yb = vb + PH;                // (b, n, d) bf16

  qkv_gemm_relu_split<<<dim3(3 * DMODEL / 128, BATCH * SEQ / 128), 256, 0, stream>>>(
      x, wqkv, qb, kb, vb);
  linattn_chunked<<<dim3(BATCH * NHEADS), 256, 0, stream>>>(qb, kb, vb, yb);
  out_gemm<<<dim3(DMODEL / 128, BATCH * SEQ / 128), 256, 0, stream>>>(yb, wo, out);
}
